// AllegroLayer_9285719294280
// MI455X (gfx1250) — matrix operand order repaired and checked by reading, not compiled, not run on hardware
//
#include <hip/hip_runtime.h>

typedef __bf16 bf16_t;
typedef bf16_t v16bf __attribute__((ext_vector_type(16)));
typedef float  v8f   __attribute__((ext_vector_type(8)));

#define NSC  128
#define TFD  144

// workspace layout (bytes)
#define CNT_OFF   (524288)   // node_Y occupies [0, N*12*4) padded region
#define PACK_OFF  (655360)

// packed-weight fragment ids (each fragment = 32 lanes * 16 bf16 = 512 bf16)
#define FR_ENV   0   // 4
#define FR_MLP1  4   // 8 ntiles * 5 ktiles
#define FR_MLP2  44  // 8 ntiles * 4 ktiles
#define FR_TENV  76  // 4
#define FR_LIN0  80  // 2
#define FR_LIN1  82  // 2
#define FR_LIN2  84  // 2
#define N_FRAGS  86

#define WMMA(A,B,C) __builtin_amdgcn_wmma_f32_16x16x32_bf16(false,(A),false,(B),(short)0,(C),false,false)

static __device__ __forceinline__ v16bf ldfrag(const bf16_t* pack, int f) {
  int lane = threadIdx.x & 31;
  const uint4* p = (const uint4*)(pack + ((size_t)(f * 32 + lane) << 4));
  union { uint4 u[2]; v16bf v; } cv;
  cv.u[0] = p[0]; cv.u[1] = p[1];
  return cv.v;
}

// A fragment (16x32 bf16, M=rows of tile) from row-major f32, row pointer at col k0.
// ISA layout: lane half h, VGPR0-3 hold K = 8h..8h+7, VGPR4-7 hold K = 16+8h..16+8h+7.
static __device__ __forceinline__ v16bf a_from_f32(const float* base_k0) {
  const int half = (threadIdx.x >> 4) & 1;
  const float* p = base_k0 + half * 8;
  float4 q0 = *(const float4*)(p);
  float4 q1 = *(const float4*)(p + 4);
  float4 q2 = *(const float4*)(p + 16);
  float4 q3 = *(const float4*)(p + 20);
  v16bf a;
  a[0]=(bf16_t)q0.x;  a[1]=(bf16_t)q0.y;  a[2]=(bf16_t)q0.z;  a[3]=(bf16_t)q0.w;
  a[4]=(bf16_t)q1.x;  a[5]=(bf16_t)q1.y;  a[6]=(bf16_t)q1.z;  a[7]=(bf16_t)q1.w;
  a[8]=(bf16_t)q2.x;  a[9]=(bf16_t)q2.y;  a[10]=(bf16_t)q2.z; a[11]=(bf16_t)q2.w;
  a[12]=(bf16_t)q3.x; a[13]=(bf16_t)q3.y; a[14]=(bf16_t)q3.z; a[15]=(bf16_t)q3.w;
  return a;
}

static __device__ __forceinline__ v16bf a_from_bf16(const bf16_t* base_k0) {
  const int half = (threadIdx.x >> 4) & 1;
  const bf16_t* p = base_k0 + half * 8;
  union { uint4 u[2]; v16bf v; } cv;
  cv.u[0] = *(const uint4*)(p);
  cv.u[1] = *(const uint4*)(p + 16);
  return cv.v;
}

// ---------------- prep kernels ----------------

__global__ void k_zero(float* nodeY, float* cnt, const int* pN) {
  int N = *pN;
  int total = N * 13;
  for (int i = blockIdx.x * blockDim.x + threadIdx.x; i < total; i += gridDim.x * blockDim.x) {
    if (i < N * 12) nodeY[i] = 0.f; else cnt[i - N * 12] = 0.f;
  }
}

__global__ void k_scatter(const float* __restrict__ ang, const int* __restrict__ eidx,
                          float* nodeY, float* cnt, int E) {
  int e = blockIdx.x * blockDim.x + threadIdx.x;
  if (e >= E) return;
  int s = eidx[2 * e];
  #pragma unroll
  for (int d = 0; d < 9; ++d) atomicAdd(&nodeY[(size_t)s * 12 + d], ang[(size_t)e * 9 + d]);
  atomicAdd(&cnt[s], 1.f);
}

__global__ void k_norm(float* nodeY, const float* cnt, const int* pN) {
  int N = *pN;
  for (int nid = blockIdx.x * blockDim.x + threadIdx.x; nid < N; nid += gridDim.x * blockDim.x) {
    float m = cnt[nid]; m = (m > 1.f) ? m : 1.f;
    float inv = 1.f / m;
    #pragma unroll
    for (int d = 0; d < 9; ++d) nodeY[(size_t)nid * 12 + d] *= inv;
  }
}

// Pack weight matrices (row-major K x N f32) into B-fragment order.
// B layout (32x16 tile): lane n, half h: elem j<8 = B[k0 + 8h + j][n], elem j>=8 = B[k0 + 16 + 8h + (j-8)][n].
__global__ void k_pack(const float* __restrict__ envW, const float* __restrict__ w1,
                       const float* __restrict__ w2, const float* __restrict__ tenvW,
                       const float* __restrict__ l0, const float* __restrict__ l1,
                       const float* __restrict__ l2, bf16_t* __restrict__ out) {
  int f = blockIdx.x;
  int lane = threadIdx.x;
  const float* src; int K, N, kt, nt;
  if (f < 4)        { src = envW;  K = 128; N = 16;  kt = f;       nt = 0; }
  else if (f < 44)  { int g = f - 4;  src = w1;   K = 144; N = 128; nt = g / 5; kt = g % 5; }
  else if (f < 76)  { int g = f - 44; src = w2;   K = 128; N = 128; nt = g / 4; kt = g % 4; }
  else if (f < 80)  { src = tenvW; K = 128; N = 16; kt = f - 76; nt = 0; }
  else if (f < 82)  { src = l0;    K = 48;  N = 16; kt = f - 80; nt = 0; }
  else if (f < 84)  { src = l1;    K = 64;  N = 16; kt = f - 82; nt = 0; }
  else              { src = l2;    K = 64;  N = 16; kt = f - 84; nt = 0; }
  int n = lane & 15, half = lane >> 4;
  bf16_t* dst = out + ((size_t)(f * 32 + lane) << 4);
  int ng = nt * 16 + n;
  #pragma unroll
  for (int j = 0; j < 16; ++j) {
    int k = kt * 32 + (j < 8 ? half * 8 + j : 16 + half * 8 + (j - 8));
    float v = (k < K) ? src[(size_t)k * N + ng] : 0.f;
    dst[j] = (bf16_t)v;
  }
}

// ---------------- fused main kernel ----------------
// 64 threads = 2 waves; each wave owns a 16-edge tile.

__global__ __launch_bounds__(64) void k_main(
    const float* __restrict__ accum, const float* __restrict__ tfeat,
    const float* __restrict__ tpw,   const float* __restrict__ envb,
    const float* __restrict__ b1,    const float* __restrict__ b2,
    const float* __restrict__ tenvb, const int* __restrict__ eidx,
    const float* __restrict__ nodeY, const bf16_t* __restrict__ pack,
    float* __restrict__ outS, float* __restrict__ outT, int E)
{
  __shared__ __align__(16) float  sEY [2][16][12];   // gathered node_Y per edge
  __shared__ __align__(16) float  sENV[2][16][16];   // env_w, later env_out
  __shared__ __align__(16) bf16_t sF  [2][5][16][64];// TP features for current l3
  __shared__ __align__(16) float  sNB [2][16][144];  // new_blocks (pre-envelope)
  __shared__ __align__(16) float  sACT[2][16][128];  // h, later updated_scalars

  const int tid  = threadIdx.x;
  const int w    = tid >> 5;
  const int lane = tid & 31;
  const int n    = lane & 15;
  const int half = lane >> 4;
  const int tileBase = blockIdx.x * 32 + w * 16;

  // gather eY rows for the 16 edges of this wave's tile
  if (lane < 16) {
    int eg = tileBase + lane; if (eg >= E) eg = E - 1;
    int srcn = eidx[2 * eg];
    const float* ny = nodeY + (size_t)srcn * 12;
    #pragma unroll
    for (int d = 0; d < 9; ++d) sEY[w][lane][d] = ny[d];
  }

  // A fragments of accumulated_scalars (reused for env GEMM and mlp1)
  int arow = tileBase + n; if (arow >= E) arow = E - 1;
  const float* ap = accum + (size_t)arow * NSC;
  v16bf Aacc[4];
  #pragma unroll
  for (int kt = 0; kt < 4; ++kt) Aacc[kt] = a_from_f32(ap + kt * 32);

  __syncthreads();

  // ---- env_w = accum @ env_W + env_b ----
  {
    v16bf Be[4];
    #pragma unroll
    for (int kt = 0; kt < 4; ++kt) Be[kt] = ldfrag(pack, FR_ENV + kt);
    v8f c = {};
    #pragma unroll
    for (int kt = 0; kt < 4; ++kt) c = WMMA(Aacc[kt], Be[kt], c);
    float bias = envb[n];
    #pragma unroll
    for (int r = 0; r < 8; ++r) sENV[w][r + 8 * half][n] = c[r] + bias;
  }
  __syncthreads();

  // ---- tensor product (per-lane sparse real-CG) + lin (WMMA) per output l ----
  const int cch = n;
  const float tw0  = tpw[0*16+cch],  tw1  = tpw[1*16+cch],  tw2  = tpw[2*16+cch];
  const float tw3  = tpw[3*16+cch],  tw4  = tpw[4*16+cch],  tw6  = tpw[6*16+cch];
  const float tw7  = tpw[7*16+cch],  tw10 = tpw[10*16+cch], tw11 = tpw[11*16+cch];
  const float tw14 = tpw[14*16+cch], tw16 = tpw[16*16+cch];

  const float RS2 = 0.70710678118f, RS3 = 0.57735026919f, RS5 = 0.44721359550f,
              RS6 = 0.40824829046f, HRS2 = 0.35355339059f, HRS6 = 0.20412414523f,
              SQ35 = 0.77459666924f, SQ127 = 1.30930734142f;

  #pragma unroll
  for (int l3 = 0; l3 < 3; ++l3) {
    #pragma unroll
    for (int j = 0; j < 8; ++j) {
      const int e = 2 * j + half;
      int eg = tileBase + e; if (eg >= E) eg = E - 1;
      const float* tf = tfeat + (size_t)eg * TFD;
      const float ew = sENV[w][e][cch];
      const float V0 = tf[cch] * ew;
      float V1[3], V2[5], Y1[3], Y2[5];
      #pragma unroll
      for (int m = 0; m < 3; ++m) V1[m] = tf[16 + m * 16 + cch] * ew;
      #pragma unroll
      for (int m = 0; m < 5; ++m) V2[m] = tf[64 + m * 16 + cch] * ew;
      const float Y0 = sEY[w][e][0];
      #pragma unroll
      for (int m = 0; m < 3; ++m) Y1[m] = sEY[w][e][1 + m];
      #pragma unroll
      for (int m = 0; m < 5; ++m) Y2[m] = sEY[w][e][4 + m];

      if (l3 == 0) {
        float f0 = V0 * Y0 * tw0;
        float f1 = RS3 * (V1[0]*Y1[0] + V1[1]*Y1[1] + V1[2]*Y1[2]) * tw4;
        float f2 = RS5 * (V2[0]*Y2[0]+V2[1]*Y2[1]+V2[2]*Y2[2]+V2[3]*Y2[3]+V2[4]*Y2[4]) * tw14;
        sF[w][0][e][cch]      = (bf16_t)f0;
        sF[w][0][e][16 + cch] = (bf16_t)f1;
        sF[w][0][e][32 + cch] = (bf16_t)f2;
        sF[w][0][e][48 + cch] = (bf16_t)0.f;   // pad 48->64
      } else if (l3 == 1) {
        float MY[3][3], MV[3][3];  // symmetric-traceless matrix from l=2 components
        MY[0][0] = -Y2[2]*RS6 - Y2[4]*RS2; MY[0][1] = Y2[1]*RS2;      MY[0][2] = Y2[0]*RS2;
        MY[1][0] =  Y2[1]*RS2;             MY[1][1] = 2.f*Y2[2]*RS6;  MY[1][2] = Y2[3]*RS2;
        MY[2][0] =  Y2[0]*RS2;             MY[2][1] = Y2[3]*RS2;      MY[2][2] = -Y2[2]*RS6 + Y2[4]*RS2;
        MV[0][0] = -V2[2]*RS6 - V2[4]*RS2; MV[0][1] = V2[1]*RS2;      MV[0][2] = V2[0]*RS2;
        MV[1][0] =  V2[1]*RS2;             MV[1][1] = 2.f*V2[2]*RS6;  MV[1][2] = V2[3]*RS2;
        MV[2][0] =  V2[0]*RS2;             MV[2][1] = V2[3]*RS2;      MV[2][2] = -V2[2]*RS6 + V2[4]*RS2;
        #pragma unroll
        for (int k = 0; k < 3; ++k) {
          float s0 = V0 * Y1[k] * tw1;
          float s1 = V1[k] * Y0 * tw3;
          float s2 = SQ35 * (MY[k][0]*V1[0] + MY[k][1]*V1[1] + MY[k][2]*V1[2]) * tw7;
          float s3 = SQ35 * (MV[k][0]*Y1[0] + MV[k][1]*Y1[1] + MV[k][2]*Y1[2]) * tw11;
          sF[w][k][e][cch]      = (bf16_t)s0;
          sF[w][k][e][16 + cch] = (bf16_t)s1;
          sF[w][k][e][32 + cch] = (bf16_t)s2;
          sF[w][k][e][48 + cch] = (bf16_t)s3;
        }
      } else {
        float c112[5], c222[5];
        c112[0] = (V1[2]*Y1[0] + V1[0]*Y1[2]) * RS2;
        c112[1] = (V1[0]*Y1[1] + V1[1]*Y1[0]) * RS2;
        c112[2] = (2.f*V1[1]*Y1[1] - V1[2]*Y1[2] - V1[0]*Y1[0]) * RS6;
        c112[3] = (V1[1]*Y1[2] + V1[2]*Y1[1]) * RS2;
        c112[4] = (V1[2]*Y1[2] - V1[0]*Y1[0]) * RS2;
        c222[0] = -(V2[0]*Y2[2]+V2[2]*Y2[0])*RS6 + (V2[1]*Y2[3]+V2[3]*Y2[1])*HRS2;
        c222[1] =  (V2[0]*Y2[3]+V2[3]*Y2[0])*HRS2 + (V2[1]*Y2[2]+V2[2]*Y2[1])*HRS6
                  -(V2[1]*Y2[4]+V2[4]*Y2[1])*HRS2;
        c222[2] = -V2[0]*Y2[0]*RS6 + V2[1]*Y2[1]*HRS6 + V2[2]*Y2[2]*RS6
                  +V2[3]*Y2[3]*HRS6 - V2[4]*Y2[4]*RS6;
        c222[3] =  (V2[0]*Y2[1]+V2[1]*Y2[0])*HRS2 + (V2[2]*Y2[3]+V2[3]*Y2[2])*HRS6
                  +(V2[3]*Y2[4]+V2[4]*Y2[3])*HRS2;
        c222[4] = -V2[1]*Y2[1]*HRS2 - (V2[2]*Y2[4]+V2[4]*Y2[2])*RS6 + V2[3]*Y2[3]*HRS2;
        #pragma unroll
        for (int k = 0; k < 5; ++k) {
          float s0 = V0 * Y2[k] * tw2;
          float s1 = c112[k] * tw6;
          float s2 = V2[k] * Y0 * tw10;
          float s3 = SQ127 * c222[k] * tw16;
          sF[w][k][e][cch]      = (bf16_t)s0;
          sF[w][k][e][16 + cch] = (bf16_t)s1;
          sF[w][k][e][32 + cch] = (bf16_t)s2;
          sF[w][k][e][48 + cch] = (bf16_t)s3;
        }
      }
    }
    __syncthreads();

    // lin: (16 edges x 64) @ (64 x 16) per m-row, WMMA
    const int nrows = 2 * l3 + 1;
    const int off   = (l3 == 0) ? 0 : (l3 == 1 ? 16 : 64);
    const int fb    = (l3 == 0) ? FR_LIN0 : (l3 == 1 ? FR_LIN1 : FR_LIN2);
    // B fragments are invariant across the m-rows: load once.
    v16bf B0 = ldfrag(pack, fb + 0);
    v16bf B1 = ldfrag(pack, fb + 1);
    for (int k = 0; k < nrows; ++k) {
      v16bf a0 = a_from_bf16(&sF[w][k][n][0]);
      v16bf a1 = a_from_bf16(&sF[w][k][n][32]);
      v8f cc = {};
      cc = WMMA(a0, B0, cc);
      cc = WMMA(a1, B1, cc);
      #pragma unroll
      for (int r = 0; r < 8; ++r) sNB[w][r + 8 * half][off + k * 16 + n] = cc[r];
    }
    __syncthreads();
  }

  // ---- mlp1: [accum | invariants] (K padded to 160) -> silu ----
  v16bf A4;
  #pragma unroll
  for (int jj = 0; jj < 8; ++jj) {
    A4[jj]     = (bf16_t)sNB[w][n][half * 8 + jj];  // invariants = new_blocks[0][:,0,:]
    A4[8 + jj] = (bf16_t)0.f;                       // zero pad 144->160
  }
  #pragma unroll
  for (int nt = 0; nt < 8; ++nt) {
    // batch all 5 B fragments for this N-tile, then run the WMMA chain
    v16bf Bf[5];
    #pragma unroll
    for (int kt = 0; kt < 5; ++kt) Bf[kt] = ldfrag(pack, FR_MLP1 + nt * 5 + kt);
    v8f cc = {};
    #pragma unroll
    for (int kt = 0; kt < 4; ++kt) cc = WMMA(Aacc[kt], Bf[kt], cc);
    cc = WMMA(A4, Bf[4], cc);
    float bias = b1[nt * 16 + n];
    #pragma unroll
    for (int r = 0; r < 8; ++r) {
      float x = cc[r] + bias;
      sACT[w][r + 8 * half][nt * 16 + n] = x / (1.f + __expf(-x));  // silu
    }
  }
  __syncthreads();

  // ---- mlp2: updated_scalars = h @ W2 + b2 ----
  {
    v16bf Ah[4];
    #pragma unroll
    for (int kt = 0; kt < 4; ++kt) Ah[kt] = a_from_f32(&sACT[w][n][kt * 32]);
    #pragma unroll
    for (int nt = 0; nt < 8; ++nt) {
      v16bf Bf[4];
      #pragma unroll
      for (int kt = 0; kt < 4; ++kt) Bf[kt] = ldfrag(pack, FR_MLP2 + nt * 4 + kt);
      v8f cc = {};
      #pragma unroll
      for (int kt = 0; kt < 4; ++kt) cc = WMMA(Ah[kt], Bf[kt], cc);
      float bias = b2[nt * 16 + n];
      #pragma unroll
      for (int r = 0; r < 8; ++r) {
        float v = cc[r] + bias;
        int row = tileBase + r + 8 * half;
        if (row < E) outS[(size_t)row * NSC + nt * 16 + n] = v;
        sACT[w][r + 8 * half][nt * 16 + n] = v;
      }
    }
  }
  __syncthreads();

  // ---- tenv: env_out = updated_scalars @ tenv_W + tenv_b ----
  {
    v16bf Au[4];
    #pragma unroll
    for (int kt = 0; kt < 4; ++kt) Au[kt] = a_from_f32(&sACT[w][n][kt * 32]);
    v16bf Bt[4];
    #pragma unroll
    for (int kt = 0; kt < 4; ++kt) Bt[kt] = ldfrag(pack, FR_TENV + kt);
    v8f cc = {};
    #pragma unroll
    for (int kt = 0; kt < 4; ++kt) cc = WMMA(Au[kt], Bt[kt], cc);
    float bias = tenvb[n];
    #pragma unroll
    for (int r = 0; r < 8; ++r) sENV[w][r + 8 * half][n] = cc[r] + bias;
  }
  __syncthreads();

  // ---- updated_tensor = new_blocks * env_out ----
  #pragma unroll
  for (int j = 0; j < 8; ++j) {
    const int e = 2 * j + half;
    const int row = tileBase + e;
    if (row < E) {
      const float eo = sENV[w][e][cch];
      float* op = outT + (size_t)row * TFD;
      #pragma unroll
      for (int m9 = 0; m9 < 9; ++m9) {
        int d = m9 * 16 + cch;
        op[d] = sNB[w][e][d] * eo;
      }
    }
  }
}

// ---------------- host ----------------

extern "C" void kernel_launch(void* const* d_in, const int* in_sizes, int n_in,
                              void* d_out, int out_size, void* d_ws, size_t ws_size,
                              hipStream_t stream) {
  const float* accum = (const float*)d_in[0];
  const float* tfeat = (const float*)d_in[1];
  const float* ang   = (const float*)d_in[2];
  const float* tpw   = (const float*)d_in[3];
  const float* envW  = (const float*)d_in[4];
  const float* envb  = (const float*)d_in[5];
  const float* w1    = (const float*)d_in[6];
  const float* b1    = (const float*)d_in[7];
  const float* w2    = (const float*)d_in[8];
  const float* b2    = (const float*)d_in[9];
  const float* tenvW = (const float*)d_in[10];
  const float* tenvb = (const float*)d_in[11];
  const int*   eidx  = (const int*)d_in[12];
  const int*   pN    = (const int*)d_in[13];
  const float* l0    = (const float*)d_in[14];
  const float* l1    = (const float*)d_in[15];
  const float* l2    = (const float*)d_in[16];

  const int E = in_sizes[0] / NSC;

  float*  nodeY = (float*)d_ws;
  float*  cnt   = (float*)((char*)d_ws + CNT_OFF);
  bf16_t* packW = (bf16_t*)((char*)d_ws + PACK_OFF);

  float* outS = (float*)d_out;
  float* outT = outS + (size_t)E * NSC;

  k_zero<<<256, 256, 0, stream>>>(nodeY, cnt, pN);
  k_pack<<<N_FRAGS, 32, 0, stream>>>(envW, w1, w2, tenvW, l0, l1, l2, packW);
  k_scatter<<<(E + 255) / 256, 256, 0, stream>>>(ang, eidx, nodeY, cnt, E);
  k_norm<<<128, 256, 0, stream>>>(nodeY, cnt, pN);
  k_main<<<(E + 31) / 32, 64, 0, stream>>>(accum, tfeat, tpw, envb, b1, b2, tenvb,
                                           eidx, nodeY, packW, outS, outT, E);
}
